// Encoder_52819507806279
// MI455X (gfx1250) — matrix operand order repaired and checked by reading, not compiled, not run on hardware
//
#include <hip/hip_runtime.h>

typedef __attribute__((ext_vector_type(16))) __bf16 v16bf;
typedef __attribute__((ext_vector_type(8)))  float  v8f;

union BFrag { v16bf v; unsigned short u[16]; };

#define BM 64
#define BN 128
#define BK 64
#define BKP 72    // LDS row stride in ushorts: 144B rows -> 16B-aligned b128 stores,
                  // 36-dword stride -> conflict-free strided fragment reads
#define KCPAD 192 // conv im2col K (147) padded to 3*BK

__device__ __forceinline__ unsigned short f2bf(float f) {
  union { float f; unsigned u; } x; x.f = f;
  unsigned r = x.u + 0x7fffu + ((x.u >> 16) & 1u);   // round-to-nearest-even
  return (unsigned short)(r >> 16);
}

// ---------------------------------------------------------------------------
// Tiled bf16 WMMA GEMM:  out[M,N] = A[M,K] * Bw[N,K]^T + bias
// 256 threads = 8 waves (2x4); each wave owns a 32x32 tile.
// BK=64 -> 8 v_wmma per wave per K-step.
// Double-buffered LDS filled by GLOBAL_LOAD_ASYNC_TO_LDS_B128 (ASYNCcnt),
// overlapped with compute; one s_wait_asynccnt + barrier per K-step.
// rowmap (optional) gathers A rows -> fuses the per-sample token permutation.
// BF16OUT: apply PReLU(alpha) and store bf16; else store f32 with row stride ldd.
// ---------------------------------------------------------------------------
template <bool BF16OUT>
__global__ __launch_bounds__(256)
void gemm_bf16(const unsigned short* __restrict__ A,
               const unsigned short* __restrict__ Bw,
               const float* __restrict__ bias,
               const float* __restrict__ alpha,
               float* __restrict__ outF,
               unsigned short* __restrict__ outH,
               const int* __restrict__ rowmap,
               int M, int N, int K, int ldd)
{
  __shared__ alignas(128) unsigned short As[2][BM][BKP];
  __shared__ alignas(128) unsigned short Bs[2][BN][BKP];

  const int t    = threadIdx.x;
  const int lane = t & 31;           // wave32
  const int wid  = t >> 5;           // 0..7
  const int wm   = (wid >> 2) * 32;  // 0 / 32
  const int wn   = (wid & 3) * 32;   // 0..96
  const int n0   = blockIdx.x * BN;
  const int m0   = blockIdx.y * BM;

  // A staging: thread -> (row = t/4, 16 ushorts at kc=(t%4)*16) : 2x b128
  const int arow = t >> 2;
  const int akc  = (t & 3) * 16;
  int asrc = m0 + arow;
  if (rowmap) asrc = rowmap[asrc];
  const unsigned short* aptr = A + (size_t)asrc * K + akc;

  // B staging: thread -> (row = t/2, 32 ushorts at kc=(t%2)*32) : 4x b128
  const int brow = t >> 1;
  const int bkc  = (t & 1) * 32;
  const unsigned short* bptr = Bw + (size_t)(n0 + brow) * K + bkc;

  // LDS byte offsets of this thread's staging slots (generic ptr low 32 bits
  // = wave-relative LDS address per the gfx1250 aperture mapping)
  const unsigned ldsA0 = (unsigned)(size_t)&As[0][arow][akc];
  const unsigned ldsB0 = (unsigned)(size_t)&Bs[0][brow][bkc];

  // async global->LDS staging of one BMxBK / BNxBK tile pair into buffer q
  auto stage_async = [&](int k, int q) {
    const unsigned long long ga = (unsigned long long)(aptr + k);
    const unsigned long long gb = (unsigned long long)(bptr + k);
    const unsigned la = ldsA0 + (unsigned)q * (unsigned)sizeof(As[0]);
    const unsigned lb = ldsB0 + (unsigned)q * (unsigned)sizeof(Bs[0]);
#pragma unroll
    for (int i = 0; i < 2; ++i)
      asm volatile("global_load_async_to_lds_b128 %0, %1, off"
                   :: "v"(la + 16u * i), "v"(ga + 16ull * i) : "memory");
#pragma unroll
    for (int i = 0; i < 4; ++i)
      asm volatile("global_load_async_to_lds_b128 %0, %1, off"
                   :: "v"(lb + 16u * i), "v"(gb + 16ull * i) : "memory");
  };
  auto wait_async = [&]() {
    asm volatile("s_wait_asynccnt 0x0" ::: "memory");
  };

  v8f cc[2][2] = {};

  const int lm  = lane & 15;
  const int hi  = lane >> 4;   // half-wave select
  const int kbA = hi * 8;      // A: lanes 16-31 hold K groups +8 (ISA 7.12.2)
  const int kbB = hi * 8;      // B: same K order as A (lanes 16-31 hold K groups +8)

  // prologue: stage tile 0 into buffer 0
  stage_async(0, 0);
  wait_async();
  __syncthreads();

  int p = 0;
  for (int k0 = 0; k0 < K; k0 += BK) {
    const bool last = (k0 + BK >= K);
    if (!last) {
      if (k0 + 2 * BK < K) {
        __builtin_prefetch(aptr + k0 + 2 * BK, 0, 1);   // global_prefetch_b8
        __builtin_prefetch(bptr + k0 + 2 * BK, 0, 1);
      }
      stage_async(k0 + BK, p ^ 1);   // DMA next tile while we compute this one
    }

#pragma unroll
    for (int s = 0; s < 2; ++s) {        // two 16x16x32 sub-steps of the BK=64 tile
      const int ks = s * 32;
      BFrag a[2], b[2];
#pragma unroll
      for (int v = 0; v < 8; ++v) {
        int kp = ks + ((v & 4) ? 16 : 0) + kbA + (v & 3) * 2;
        a[0].u[2 * v]     = As[p][wm + lm][kp];
        a[0].u[2 * v + 1] = As[p][wm + lm][kp + 1];
        a[1].u[2 * v]     = As[p][wm + 16 + lm][kp];
        a[1].u[2 * v + 1] = As[p][wm + 16 + lm][kp + 1];
      }
#pragma unroll
      for (int e = 0; e < 16; ++e) {
        b[0].u[e] = Bs[p][wn + lm][ks + kbB + e + ((e & 8) ? 8 : 0)];
        b[1].u[e] = Bs[p][wn + 16 + lm][ks + kbB + e + ((e & 8) ? 8 : 0)];
      }
#pragma unroll
      for (int i = 0; i < 2; ++i)
#pragma unroll
        for (int j = 0; j < 2; ++j)
          cc[i][j] = __builtin_amdgcn_wmma_f32_16x16x32_bf16(
              false, a[i].v, false, b[j].v, (short)0, cc[i][j], false, false);
    }

    if (!last) {
      wait_async();        // async tile landed in LDS
      __syncthreads();     // single barrier per K-step
      p ^= 1;
    }
  }

  // epilogue (D layout: M = v + hi*8, N = lane%16)
  const float al = BF16OUT ? alpha[0] : 0.f;
#pragma unroll
  for (int i = 0; i < 2; ++i) {
#pragma unroll
    for (int j = 0; j < 2; ++j) {
      int col  = n0 + wn + 16 * j + lm;
      float bv = bias[col];
#pragma unroll
      for (int v = 0; v < 8; ++v) {
        int row   = m0 + wm + 16 * i + hi * 8 + v;
        float val = cc[i][j][v] + bv;
        if (BF16OUT) {
          val = val > 0.f ? val : al * val;
          outH[(size_t)row * N + col] = f2bf(val);
        } else {
          outF[(size_t)row * ldd + col] = val;
        }
      }
    }
  }
}

// ---------------------------------------------------------------------------
// Helper kernels
// ---------------------------------------------------------------------------
__global__ void k_f32_to_bf16(const float* __restrict__ in,
                              unsigned short* __restrict__ out, int n) {
  int i = blockIdx.x * blockDim.x + threadIdx.x;
  if (i < n) out[i] = f2bf(in[i]);
}

__global__ void k_pad_w0(const float* __restrict__ w0,
                         unsigned short* __restrict__ out) {
  int i = blockIdx.x * blockDim.x + threadIdx.x;   // 1024*KCPAD
  if (i >= 1024 * KCPAD) return;
  int d = i / KCPAD, j = i % KCPAD;
  out[i] = (j < 147) ? f2bf(w0[d * 147 + j]) : (unsigned short)0;
}

// x[B=128,3,112,112] -> patches[m = b*256 + ph*16 + pw][j = cin*49 + kh*7 + kw], K padded
__global__ void k_im2col(const float* __restrict__ x,
                         unsigned short* __restrict__ out) {
  int i = blockIdx.x * blockDim.x + threadIdx.x;   // 32768*KCPAD
  if (i >= 32768 * KCPAD) return;
  int m = i / KCPAD, j = i % KCPAD;
  unsigned short v = 0;
  if (j < 147) {
    int cin = j / 49, r = j % 49, kh = r / 7, kw = r % 7;
    int b = m >> 8, p = m & 255, ph = p >> 4, pw = p & 15;
    v = f2bf(x[(((size_t)b * 3 + cin) * 112 + (ph * 7 + kh)) * 112 + (pw * 7 + kw)]);
  }
  out[i] = v;
}

__global__ void k_zero(float* __restrict__ p, int n) {
  int i = blockIdx.x * blockDim.x + threadIdx.x;
  if (i < n) p[i] = 0.f;
}

// per-channel sum / sumsq over M rows of [M,1024] f32 (batch-stat BN)
__global__ __launch_bounds__(256)
void k_stats(const float* __restrict__ xf, float* __restrict__ sums) {
  __shared__ float s1[256], s2[256];
  int ch = blockIdx.x * 64 + (threadIdx.x & 63);
  int rs = threadIdx.x >> 6;
  int r0 = blockIdx.y * 256;
  float a = 0.f, b = 0.f;
  for (int r = r0 + rs; r < r0 + 256; r += 4) {
    float v = xf[(size_t)r * 1024 + ch];
    a += v; b += v * v;
  }
  s1[threadIdx.x] = a; s2[threadIdx.x] = b;
  __syncthreads();
  if (threadIdx.x < 64) {
    int t = threadIdx.x;
    float ta = s1[t] + s1[t + 64] + s1[t + 128] + s1[t + 192];
    float tb = s2[t] + s2[t + 64] + s2[t + 128] + s2[t + 192];
    atomicAdd(&sums[ch], ta);
    atomicAdd(&sums[1024 + ch], tb);
  }
}

__global__ void k_bn_coef(const float* __restrict__ sums, const float* __restrict__ g,
                          const float* __restrict__ bt, float* __restrict__ coef, int M) {
  int c = blockIdx.x * blockDim.x + threadIdx.x;
  if (c >= 1024) return;
  float inv  = 1.f / (float)M;
  float mean = sums[c] * inv;
  float var  = sums[1024 + c] * inv - mean * mean;
  float sc   = g[c] * rsqrtf(var + 1e-5f);
  coef[c]        = sc;
  coef[1024 + c] = bt[c] - mean * sc;
}

__global__ void k_bn_prelu(const float* __restrict__ xf, const float* __restrict__ coef,
                           const float* __restrict__ p, unsigned short* __restrict__ out,
                           int total) {
  int i = blockIdx.x * blockDim.x + threadIdx.x;
  if (i >= total) return;
  int d   = i & 1023;
  float y = xf[i] * coef[d] + coef[1024 + d];
  float pp = p[0];
  y = y > 0.f ? y : pp * y;
  out[i] = f2bf(y);
}

__global__ void k_bn_prelu_pos(const float* __restrict__ xf, const float* __restrict__ coef,
                               const float* __restrict__ p, const float* __restrict__ pos,
                               const float* __restrict__ am, unsigned short* __restrict__ out,
                               int total) {
  int i = blockIdx.x * blockDim.x + threadIdx.x;
  if (i >= total) return;
  int d  = i & 1023;
  int hw = (i >> 10) & 255;
  float y  = xf[i] * coef[d] + coef[1024 + d];
  float pp = p[0];
  y = y > 0.f ? y : pp * y;              // PReLU(p3)
  y += pos[hw * 1024 + d];               // + positional embedding
  float aa = am[0];
  y = y > 0.f ? y : aa * y;              // PReLU(am) -> mixer input
  out[i] = f2bf(y);
}

// out row m gathers mixer-input row (b*256 + perm[b,i])
__global__ void k_rowmap(const int* __restrict__ perm, int* __restrict__ rs, int n) {
  int i = blockIdx.x * blockDim.x + threadIdx.x;
  if (i < n) rs[i] = (i & ~255) + perm[i];
}

// ---------------------------------------------------------------------------
extern "C" void kernel_launch(void* const* d_in, const int* in_sizes, int n_in,
                              void* d_out, int out_size, void* d_ws, size_t ws_size,
                              hipStream_t stream) {
  const float* x   = (const float*)d_in[0];
  const float* w0  = (const float*)d_in[1];
  const float* b0  = (const float*)d_in[2];
  const float* a0  = (const float*)d_in[3];
  const float* w[3]  = {(const float*)d_in[4], (const float*)d_in[9],  (const float*)d_in[14]};
  const float* bb[3] = {(const float*)d_in[5], (const float*)d_in[10], (const float*)d_in[15]};
  const float* g[3]  = {(const float*)d_in[6], (const float*)d_in[11], (const float*)d_in[16]};
  const float* bt[3] = {(const float*)d_in[7], (const float*)d_in[12], (const float*)d_in[17]};
  const float* pr[3] = {(const float*)d_in[8], (const float*)d_in[13], (const float*)d_in[18]};
  const float* pos = (const float*)d_in[19];
  const float* am  = (const float*)d_in[20];
  const float* wm  = (const float*)d_in[21];
  const float* bm  = (const float*)d_in[22];
  const int*  perm = (const int*)d_in[23];

  const int M = 32768, C = 1024, HID = 512, KC = KCPAD;
  const int TOT = M * C;   // 33554432

  char* base = (char*)d_ws;
  size_t off = 0;
  auto carve = [&](size_t bytes) -> void* {
    void* p = base + off;
    off = (off + bytes + 255) & ~(size_t)255;
    return p;
  };
  unsigned short* patches = (unsigned short*)carve((size_t)M * KC * 2);
  unsigned short* w0p     = (unsigned short*)carve((size_t)C * KC * 2);
  unsigned short* wb[3];
  for (int l = 0; l < 3; ++l) wb[l] = (unsigned short*)carve((size_t)C * C * 2);
  unsigned short* wmb  = (unsigned short*)carve((size_t)HID * C * 2);
  unsigned short* actH = (unsigned short*)carve((size_t)M * C * 2);
  float* gemmF  = (float*)carve((size_t)M * C * 4);
  float* sums   = (float*)carve(2048 * 4);
  float* coef   = (float*)carve(2048 * 4);
  int*   rowsrc = (int*)carve((size_t)M * 4);

  // ---- prep: bf16 conversions / im2col / permutation row map ----
  k_im2col<<<(M * KC + 255) / 256, 256, 0, stream>>>(x, patches);
  k_pad_w0<<<(C * KC + 255) / 256, 256, 0, stream>>>(w0, w0p);
  for (int l = 0; l < 3; ++l)
    k_f32_to_bf16<<<(C * C + 255) / 256, 256, 0, stream>>>(w[l], wb[l], C * C);
  k_f32_to_bf16<<<(HID * C + 255) / 256, 256, 0, stream>>>(wm, wmb, HID * C);
  k_rowmap<<<M / 256, 256, 0, stream>>>(perm, rowsrc, M);

  // ---- patch-embed conv as GEMM, fused bias + PReLU(a0) -> bf16 ----
  gemm_bf16<true><<<dim3(C / BN, M / BM), 256, 0, stream>>>(
      patches, w0p, b0, a0, nullptr, actH, nullptr, M, C, KC, 0);

  // ---- 3x (1x1 conv GEMM -> BN stats -> normalize+PReLU -> bf16) ----
  for (int l = 0; l < 3; ++l) {
    gemm_bf16<false><<<dim3(C / BN, M / BM), 256, 0, stream>>>(
        actH, wb[l], bb[l], nullptr, gemmF, nullptr, nullptr, M, C, C, C);
    k_zero<<<8, 256, 0, stream>>>(sums, 2048);
    k_stats<<<dim3(C / 64, M / 256), 256, 0, stream>>>(gemmF, sums);
    k_bn_coef<<<4, 256, 0, stream>>>(sums, g[l], bt[l], coef, M);
    if (l < 2)
      k_bn_prelu<<<TOT / 256, 256, 0, stream>>>(gemmF, coef, pr[l], actH, TOT);
    else
      k_bn_prelu_pos<<<TOT / 256, 256, 0, stream>>>(gemmF, coef, pr[l], pos, am, actH, TOT);
  }

  // ---- mixer GEMM with fused A-row gather (token permutation) + bias ----
  gemm_bf16<false><<<dim3(HID / BN, M / BM), 256, 0, stream>>>(
      actH, wmb, bm, nullptr, (float*)d_out, nullptr, rowsrc, M, HID, C, HID);
}
